// RNNBinaryClassifier_74062416053006
// MI455X (gfx1250) — matrix operand order repaired and checked by reading, not compiled, not run on hardware
//
#include <hip/hip_runtime.h>
#include <hip/hip_bf16.h>

typedef __attribute__((ext_vector_type(16))) __bf16 v16bf;
typedef __attribute__((ext_vector_type(8)))  __bf16 v8bf;
typedef __attribute__((ext_vector_type(4)))  __bf16 v4bf;
typedef __attribute__((ext_vector_type(8)))  float  v8f;
typedef __attribute__((ext_vector_type(4)))  float  v4f;
typedef __attribute__((ext_vector_type(4)))  unsigned int v4u;
typedef __attribute__((ext_vector_type(8)))  int    v8i;
typedef __attribute__((ext_vector_type(4)))  int    v4i;

#define WMMA_BF16(a, b, c) \
  __builtin_amdgcn_wmma_f32_16x16x32_bf16(false, (a), false, (b), (short)0, (c), false, false)

constexpr int BATCH = 64;
constexpr int T     = 512;
constexpr int D     = 2048;
constexpr int H     = 128;

// ---------------------------------------------------------------------------
// Kernel 1: pre1[m][n] = sum_k x[m][k] * W_ih1[n][k] + (b_ih1[n] + b_hh1[n])
// M = B*T = 32768 (row = b*T + t), K = 2048, N = 128. Memory-bound (x=256MB):
// bf16 WMMA w/ f32 accumulate. 256 threads/WG; wave owns 16 rows x all 8 N-tiles.
// ---------------------------------------------------------------------------
__global__ __launch_bounds__(256)
void pre1_gemm_kernel(const float* __restrict__ x,
                      const float* __restrict__ Wih,
                      const float* __restrict__ bih,
                      const float* __restrict__ bhh,
                      float* __restrict__ pre)
{
    __shared__ __bf16 As[128][32];   // x tile, [m][k]  (A source)
    __shared__ __bf16 Bs[128][32];   // W rows, [n][k]  (B source)

    const int tid   = threadIdx.x;
    const int lane  = tid & 31;
    const int wv    = tid >> 5;
    const int mtile = blockIdx.x * 128;
    const int n_lo  = lane & 15;
    const int khalf = (lane >> 4) & 1;

    v8f acc[8];
#pragma unroll
    for (int i = 0; i < 8; ++i) acc[i] = v8f{};

    const int lrow = tid >> 1;
    const int lcol = (tid & 1) * 16;

    for (int kc = 0; kc < D / 32; ++kc) {
        const int k0 = kc * 32;
        __syncthreads();
        {
            const float* gx = x   + (size_t)(mtile + lrow) * D + k0 + lcol;
            const float* gw = Wih + (size_t)lrow * D + k0 + lcol;
#pragma unroll
            for (int q = 0; q < 4; ++q) {
                v4f xa = *(const v4f*)(gx + q * 4);
                v4f wa = *(const v4f*)(gw + q * 4);
                v4bf xb, wb;
#pragma unroll
                for (int e = 0; e < 4; ++e) {
                    xb[e] = (__bf16)xa[e];
                    wb[e] = (__bf16)wa[e];
                }
                *(v4bf*)&As[lrow][lcol + q * 4] = xb;
                *(v4bf*)&Bs[lrow][lcol + q * 4] = wb;
            }
        }
        __syncthreads();

        const int am   = 16 * wv + n_lo;
        const int aoff = khalf * 16;
        union { v16bf v; v8bf h[2]; } ua;
        ua.h[0] = *(const v8bf*)&As[am][aoff];
        ua.h[1] = *(const v8bf*)&As[am][aoff + 8];

        const int boff = khalf * 16;
#pragma unroll
        for (int nt = 0; nt < 8; ++nt) {
            v16bf bf = *(const v16bf*)&Bs[nt * 16 + n_lo][boff];
            acc[nt] = WMMA_BF16(ua.v, bf, acc[nt]);
        }
    }

#pragma unroll
    for (int nt = 0; nt < 8; ++nt) {
        const int n = nt * 16 + n_lo;
        const float bias = bih[n] + bhh[n];
#pragma unroll
        for (int j = 0; j < 8; ++j) {
            const int m = 16 * wv + j + khalf * 8;
            pre[(size_t)(mtile + m) * H + n] = acc[nt][j] + bias;
        }
    }
}

// ---------------------------------------------------------------------------
// TDM: async-load one step's pre-activation tile (16 rows x 128 f32, row
// stride T*H) into LDS. Descriptor per CDNA5 ISA.
// 6-arg builtin form (clang-23 / therock-10.0 headers).
// ---------------------------------------------------------------------------
__device__ __forceinline__ void tdm_load_pre_tile(const float* gsrc, unsigned lds_addr)
{
    const unsigned long long ga = (unsigned long long)(uintptr_t)gsrc;
    // Group 0: count=1 | lds_addr | global_addr[56:0] | type=2
    v4u g0 = { 1u,
               lds_addr,
               (unsigned)(ga & 0xFFFFFFFFu),
               (unsigned)((ga >> 32) & 0x01FFFFFFu) | (2u << 30) };
    // Group 1: wg_mask=0, data_size=2 (4B), no barrier/iterate/pad,
    // tensor_dim0=128, tensor_dim1=16, tile_dim0=128, tile_dim1=16,
    // tile_dim2=0, tensor_dim0_stride=T*H elements.
    v8i g1 = { (int)(2u << 16),          // data_size = 4 bytes
               (int)(128u << 16),        // tensor_dim0[15:0] << 16
               (int)(16u  << 16),        // tensor_dim0[31:16]=0 | tensor_dim1[15:0]<<16
               (int)(128u << 16),        // tensor_dim1[31:16]=0 | tile_dim0<<16
               (int)16,                  // tile_dim1=16 | tile_dim2=0
               (int)(T * H),             // tensor_dim0_stride low 32 (=65536)
               0, 0 };
    v4i g2 = {0, 0, 0, 0};
    v4i g3 = {0, 0, 0, 0};
    v8i g4 = {0, 0, 0, 0, 0, 0, 0, 0};
    __builtin_amdgcn_tensor_load_to_lds(g0, g1, g2, g3, g4, 0);
}

// ---------------------------------------------------------------------------
// Kernel 2: fused two-layer tanh-RNN scan + final FC.
// grid = 4 WGs (16 batch rows each), block = 256 (8 waves; wave owns N-tile).
// Weights register-resident as WMMA B fragments; h state bf16 in LDS (dbl buf);
// pre1 tiles DMA'd one step ahead by the Tensor Data Mover.
// ---------------------------------------------------------------------------
__device__ __forceinline__ v16bf load_a_frag(const __bf16 (*buf)[H], int kc, int lane)
{
    const int m   = lane & 15;
    const int off = ((lane >> 4) & 1) * 16;
    const int k   = kc * 32;
    union { v16bf v; v8bf h[2]; } u;
    u.h[0] = *(const v8bf*)&buf[m][k + off];
    u.h[1] = *(const v8bf*)&buf[m][k + off + 8];
    return u.v;
}

__global__ __launch_bounds__(256)
void rnn_scan_kernel(const float* __restrict__ pre1,
                     const float* __restrict__ Whh1,
                     const float* __restrict__ Wih2,
                     const float* __restrict__ Whh2,
                     const float* __restrict__ bih2,
                     const float* __restrict__ bhh2,
                     const float* __restrict__ Wfc,
                     const float* __restrict__ bfc,
                     float* __restrict__ out)
{
    __shared__ __bf16 h1b[2][16][H];
    __shared__ __bf16 h2b[2][16][H];
    __shared__ float  preb[2][16][H];   // TDM-staged pre-activations
    __shared__ float  outacc[16];

    const int tid  = threadIdx.x;
    const int lane = tid & 31;
    const int wv   = tid >> 5;
    const int b0   = blockIdx.x * 16;
    const int n    = wv * 16 + (lane & 15);
    const int hi   = (lane >> 4) & 1;
    const int kh16 = hi * 16;

    // --- preload weights as register-resident B fragments (W[n][k] layout) ---
    v16bf Bh1[4], Bi2[4], Bh2[4];
#pragma unroll
    for (int kc = 0; kc < 4; ++kc) {
        const int koff = kc * 32 + kh16;
        const float* p1 = Whh1 + (size_t)n * H + koff;
        const float* p2 = Wih2 + (size_t)n * H + koff;
        const float* p3 = Whh2 + (size_t)n * H + koff;
        v16bf f1, f2, f3;
#pragma unroll
        for (int q = 0; q < 4; ++q) {
            v4f t1 = *(const v4f*)(p1 + q * 4);
            v4f t2 = *(const v4f*)(p2 + q * 4);
            v4f t3 = *(const v4f*)(p3 + q * 4);
#pragma unroll
            for (int e = 0; e < 4; ++e) {
                f1[q * 4 + e] = (__bf16)t1[e];
                f2[q * 4 + e] = (__bf16)t2[e];
                f3[q * 4 + e] = (__bf16)t3[e];
            }
        }
        Bh1[kc] = f1; Bi2[kc] = f2; Bh2[kc] = f3;
    }
    const float bias2 = bih2[n] + bhh2[n];
    const float wfcn  = Wfc[n];

    const unsigned preb_lds0 = (unsigned)(uintptr_t)&preb[0][0][0];
    const unsigned preb_lds1 = (unsigned)(uintptr_t)&preb[1][0][0];
    const float*   pre_base  = pre1 + (size_t)b0 * T * H;   // row stride T*H

    // --- init: h(0)=0, out accumulator = b_fc; kick off TDM for t=0 ---
    for (int i = tid; i < 16 * H; i += 256) {
        ((__bf16*)h1b[0])[i] = (__bf16)0.f;
        ((__bf16*)h2b[0])[i] = (__bf16)0.f;
    }
    if (tid < 16) outacc[tid] = bfc[0];
    if (wv == 0) {
        tdm_load_pre_tile(pre_base, preb_lds0);        // tile for t=0
        __builtin_amdgcn_s_wait_tensorcnt(0);
    }
    __syncthreads();

    float h2last[8];
#pragma unroll 1
    for (int t = 0; t < T; ++t) {
        const int r = t & 1, w = r ^ 1;

        // DMA next step's tile while we compute this one (double-buffered)
        if (wv == 0 && t + 1 < T)
            tdm_load_pre_tile(pre_base + (size_t)(t + 1) * H,
                              w ? preb_lds1 : preb_lds0);

        // this step's pre-activations, staged in LDS by the TDM
        float p[8];
#pragma unroll
        for (int j = 0; j < 8; ++j)
            p[j] = preb[r][j + hi * 8][n];

        // GEMM1: h1_prev @ Whh1^T   and   GEMM3: h2_prev @ Whh2^T
        v8f a1 = v8f{}, a3 = v8f{};
#pragma unroll
        for (int kc = 0; kc < 4; ++kc)
            a1 = WMMA_BF16(load_a_frag(h1b[r], kc, lane), Bh1[kc], a1);
#pragma unroll
        for (int kc = 0; kc < 4; ++kc)
            a3 = WMMA_BF16(load_a_frag(h2b[r], kc, lane), Bh2[kc], a3);

        // h1_new = tanh(pre + recurrent); publish bf16 to LDS
#pragma unroll
        for (int j = 0; j < 8; ++j) {
            const int m = j + hi * 8;
            h1b[w][m][n] = (__bf16)tanhf(a1[j] + p[j]);
        }
        __syncthreads();   // h1_new visible to all waves

        // GEMM2: h1_new @ Wih2^T ; h2_new = tanh(. + a3 + bias2)
        v8f a2 = v8f{};
#pragma unroll
        for (int kc = 0; kc < 4; ++kc)
            a2 = WMMA_BF16(load_a_frag(h1b[w], kc, lane), Bi2[kc], a2);
#pragma unroll
        for (int j = 0; j < 8; ++j) {
            const int m = j + hi * 8;
            const float v = tanhf(a2[j] + a3[j] + bias2);
            h2b[w][m][n] = (__bf16)v;
            h2last[j] = v;
        }

        // next step's DMA must have landed before anyone passes this barrier
        if (wv == 0) __builtin_amdgcn_s_wait_tensorcnt(0);
        __syncthreads();
    }

    // --- final FC: out[b] = dot(h2_last[b,:], W_fc) + b_fc ---
#pragma unroll
    for (int j = 0; j < 8; ++j)
        atomicAdd(&outacc[j + hi * 8], h2last[j] * wfcn);
    __syncthreads();
    if (tid < 16) out[b0 + tid] = outacc[tid];
}

// ---------------------------------------------------------------------------
extern "C" void kernel_launch(void* const* d_in, const int* in_sizes, int n_in,
                              void* d_out, int out_size, void* d_ws, size_t ws_size,
                              hipStream_t stream)
{
    const float* x     = (const float*)d_in[0];
    const float* W_ih1 = (const float*)d_in[1];
    const float* W_hh1 = (const float*)d_in[2];
    const float* b_ih1 = (const float*)d_in[3];
    const float* b_hh1 = (const float*)d_in[4];
    const float* W_ih2 = (const float*)d_in[5];
    const float* W_hh2 = (const float*)d_in[6];
    const float* b_ih2 = (const float*)d_in[7];
    const float* b_hh2 = (const float*)d_in[8];
    const float* W_fc  = (const float*)d_in[9];
    const float* b_fc  = (const float*)d_in[10];

    float* pre1 = (float*)d_ws;    // (B*T) x H f32 = 16 MB

    pre1_gemm_kernel<<<(BATCH * T) / 128, 256, 0, stream>>>(x, W_ih1, b_ih1, b_hh1, pre1);
    rnn_scan_kernel<<<BATCH / 16, 256, 0, stream>>>(pre1, W_hh1, W_ih2, W_hh2,
                                                    b_ih2, b_hh2, W_fc, b_fc,
                                                    (float*)d_out);
}
